// FlashMultiHeadAttention_31379031065005
// MI455X (gfx1250) — matrix operand order repaired and checked by reading, not compiled, not run on hardware
//
#include <hip/hip_runtime.h>
#include <hip/hip_bf16.h>

typedef __attribute__((ext_vector_type(16))) __bf16       v16bf;
typedef __attribute__((ext_vector_type(8)))  float        v8f;
typedef __attribute__((ext_vector_type(4)))  float        v4f;
typedef __attribute__((ext_vector_type(2)))  float        v2f;
typedef __attribute__((ext_vector_type(8)))  unsigned int v8u;
typedef __attribute__((ext_vector_type(4)))  unsigned int v4u;
typedef __attribute__((ext_vector_type(4)))  int          v4i;

constexpr int N_TOK = 16384;
constexpr int DM    = 1024;
constexpr int NH    = 16;
constexpr int HD    = 64;

constexpr int BK  = 32;   // K step
constexpr int LDB = 72;   // padded LDS row length in bf16 elems (144B): conflict-free frag reads

// Build packed bf16 pair {hi(b),hi(a)} from two f32 with one v_perm_b32.
constexpr unsigned PERM_HI2 = 0x07060302u;

#if __has_builtin(__builtin_amdgcn_global_load_async_to_lds_b128)
#define HAVE_ASYNC_LDS 1
#else
#define HAVE_ASYNC_LDS 0
#endif

// Stage 16B from global to LDS: async DMA path (ASYNCcnt) when available, else VGPR bounce.
__device__ __forceinline__ void stage_b128(const unsigned short* g, unsigned short* s) {
#if HAVE_ASYNC_LDS
  __builtin_amdgcn_global_load_async_to_lds_b128((v4i*)g, (v4i*)s, 0, 0);
#else
  *(v4u*)s = *(const v4u*)g;
#endif
}

__device__ __forceinline__ void stage_fence() {
#if HAVE_ASYNC_LDS
#if __has_builtin(__builtin_amdgcn_s_wait_asynccnt)
  __builtin_amdgcn_s_wait_asynccnt(0);
#else
  asm volatile("s_wait_asynccnt 0" ::: "memory");
#endif
#endif
}

// Split f[0..15] (K pairs) into truncated-hi bf16 frag and exact-residual-lo bf16 frag.
__device__ __forceinline__ void split_pack(const float* f, v16bf& hi, v16bf& lo) {
  v8u uh, ul;
#pragma unroll
  for (int i = 0; i < 8; ++i) {
    const float a = f[2 * i], b = f[2 * i + 1];
    const unsigned ua = __float_as_uint(a), ub = __float_as_uint(b);
    uh[i] = __builtin_amdgcn_perm(ub, ua, PERM_HI2);
    const float la = a - __uint_as_float(ua & 0xffff0000u);  // exact
    const float lb = b - __uint_as_float(ub & 0xffff0000u);  // exact
    ul[i] = __builtin_amdgcn_perm(__float_as_uint(lb), __float_as_uint(la), PERM_HI2);
  }
  hi = __builtin_bit_cast(v16bf, uh);
  lo = __builtin_bit_cast(v16bf, ul);
}

// ---------- one-shot weight conversion: f32 -> (bf16 hi, bf16 lo), 8 elems/thread ----------
__global__ __launch_bounds__(256) void cvt_f32_bf16_hilo(const float* __restrict__ src,
                                                         unsigned short* __restrict__ hi,
                                                         unsigned short* __restrict__ lo) {
  const size_t idx = ((size_t)blockIdx.x * 256 + threadIdx.x) * 8;
  const v4f a = *(const v4f*)(src + idx);
  const v4f b = *(const v4f*)(src + idx + 4);
  float f[8];
#pragma unroll
  for (int j = 0; j < 4; ++j) { f[j] = a[j]; f[4 + j] = b[j]; }
  v4u h, l;
#pragma unroll
  for (int i = 0; i < 4; ++i) {
    const unsigned ua = __float_as_uint(f[2 * i]), ub = __float_as_uint(f[2 * i + 1]);
    h[i] = __builtin_amdgcn_perm(ub, ua, PERM_HI2);
    const float la = f[2 * i]     - __uint_as_float(ua & 0xffff0000u);
    const float lb = f[2 * i + 1] - __uint_as_float(ub & 0xffff0000u);
    l[i] = __builtin_amdgcn_perm(__float_as_uint(lb), __float_as_uint(la), PERM_HI2);
  }
  *(v4u*)(hi + idx) = h;  // 16B store (8 bf16)
  *(v4u*)(lo + idx) = l;
}

// A fragment: 16x32 (MxK) 16-bit layout; lane half p: K in [k0+8p,+8) and [k0+16+8p,+8).
__device__ __forceinline__ void load_a_frag_f32(const float* __restrict__ X, int row0,
                                                int k0, int lane, v16bf& hi, v16bf& lo) {
  const int r = lane & 15, p = lane >> 4;
  const float* q = X + (size_t)(row0 + r) * DM + k0 + p * 8;
  const v4f c0 = *(const v4f*)(q);
  const v4f c1 = *(const v4f*)(q + 4);
  const v4f c2 = *(const v4f*)(q + 16);
  const v4f c3 = *(const v4f*)(q + 20);
  float f[16];
#pragma unroll
  for (int j = 0; j < 4; ++j) { f[j] = c0[j]; f[4 + j] = c1[j]; f[8 + j] = c2[j]; f[12 + j] = c3[j]; }
  split_pack(f, hi, lo);
}

// Read one 16-elem bf16 fragment half (32B) from a staged LDS row: two ds_load_b128.
__device__ __forceinline__ v16bf ld_lds_frag(const unsigned short* q) {
  const v4u c0 = *(const v4u*)(q);
  const v4u c1 = *(const v4u*)(q + 16);
  v8u u;
#pragma unroll
  for (int j = 0; j < 4; ++j) { u[j] = c0[j]; u[4 + j] = c1[j]; }
  return __builtin_bit_cast(v16bf, u);
}

// ---------- O = X @ W^T + bias ; block = 8 waves -> 128 rows x 64 cols, B staged in LDS ----------
__global__ __launch_bounds__(256) void gemm_xwt_bias_f32(const float* __restrict__ X,
                                                         const unsigned short* __restrict__ Whi,
                                                         const unsigned short* __restrict__ Wlo,
                                                         const float* __restrict__ bias,
                                                         float* __restrict__ O) {
  // [double buffer][hi/lo][64 rows * LDB]
  __shared__ unsigned short sb[2][2][64 * LDB];

  const int tid  = threadIdx.x;
  const int lane = tid & 31;
  const int wave = tid >> 5;
  const int row0 = blockIdx.x * 128 + wave * 16;
  const int col0 = blockIdx.y * 64;
  const int p    = lane >> 4;
  const int nlo  = lane & 15;

  // staging: 256 threads cover 64 rows x 4 chunks of 8 bf16 (one b128 per operand each)
  const int sr = tid >> 2;              // strip row 0..63 (== output column - col0)
  const int sc = (tid & 3) * 8;         // k chunk offset in bf16 elems
  const unsigned short* ghi = Whi + (size_t)(col0 + sr) * DM + sc;
  const unsigned short* glo = Wlo + (size_t)(col0 + sr) * DM + sc;
  const int soff = sr * LDB + sc;

  v8f acc[4] = {};

  // prologue: stage k0 = 0 into buffer 0
  stage_b128(ghi, &sb[0][0][soff]);
  stage_b128(glo, &sb[0][1][soff]);
  stage_fence();
  __syncthreads();

  int buf = 0;
  for (int k0 = 0; k0 < DM; k0 += BK, buf ^= 1) {
    // prefetch next strip into the other buffer while computing this one
    if (k0 + BK < DM) {
      stage_b128(ghi + k0 + BK, &sb[buf ^ 1][0][soff]);
      stage_b128(glo + k0 + BK, &sb[buf ^ 1][1][soff]);
    }

    v16bf ahi, alo;
    load_a_frag_f32(X, row0, k0, lane, ahi, alo);
#pragma unroll
    for (int t = 0; t < 4; ++t) {
      const int rr = (t * 16 + nlo) * LDB + p * 8;
      const v16bf bhi = ld_lds_frag(&sb[buf][0][rr]);
      const v16bf blo = ld_lds_frag(&sb[buf][1][rr]);
      // f32 emulation: hi*hi + hi*lo + lo*hi
      acc[t] = __builtin_amdgcn_wmma_f32_16x16x32_bf16(false, ahi, false, bhi, (short)0, acc[t], false, false);
      acc[t] = __builtin_amdgcn_wmma_f32_16x16x32_bf16(false, ahi, false, blo, (short)0, acc[t], false, false);
      acc[t] = __builtin_amdgcn_wmma_f32_16x16x32_bf16(false, alo, false, bhi, (short)0, acc[t], false, false);
    }
    stage_fence();    // my staging for next buffer has landed in LDS
    __syncthreads();  // everyone done reading this buffer + staging visible block-wide
  }

  // C/D layout: VGPR i -> row (i + 8*(lane>>4)), col (lane&15)
  const int rbase = row0 + 8 * p;
#pragma unroll
  for (int t = 0; t < 4; ++t) {
    const int c = col0 + t * 16 + nlo;
    const float bv = bias[c];
#pragma unroll
    for (int i = 0; i < 8; ++i) {
      O[(size_t)(rbase + i) * DM + c] = acc[t][i] + bv;
    }
  }
}

// ---------- Per-token attention over the heads axis (16x16 scores, softmax, 16x64 out) ----------
__global__ __launch_bounds__(256) void attn_over_heads(const float* __restrict__ Qh,
                                                       const float* __restrict__ Kh,
                                                       const float* __restrict__ Vh,
                                                       float* __restrict__ Ctx) {
  __shared__ float attn_s[8][16][17];
  const int lane = threadIdx.x & 31;
  const int wave = threadIdx.x >> 5;
  const int tok  = blockIdx.x * 8 + wave;

  const float* q = Qh + (size_t)tok * DM;  // [16 heads][64 dim]
  const float* k = Kh + (size_t)tok * DM;
  const float* v = Vh + (size_t)tok * DM;

  const int m  = lane & 15;
  const int kb = (lane >> 4) * 2;  // K sub-select for 32-bit WMMA fragments

  // scores = qh @ kh^T : exact f32 via V_WMMA_F32_16X16X4_F32, K=64 in 16 steps
  v8f s = {};
#pragma unroll
  for (int ks = 0; ks < HD; ks += 4) {
    v2f a = *(const v2f*)(q + m * HD + ks + kb);   // A[m][ks+kb .. +1]
    v2f b = *(const v2f*)(k + m * HD + ks + kb);   // B = kh^T, same addressing
    s = __builtin_amdgcn_wmma_f32_16x16x4_f32(false, a, false, b, (short)0, s, false, false);
  }

  // softmax over key-head axis g (N axis of D = across 16-lane groups)
  constexpr float inv_scale = 0.125f;  // 1/sqrt(64)
#pragma unroll
  for (int i = 0; i < 8; ++i) {
    float x = s[i] * inv_scale;
    float mx = x;
#pragma unroll
    for (int d = 1; d < 16; d <<= 1) mx = fmaxf(mx, __shfl_xor(mx, d, 32));
    float e = __expf(x - mx);
    float sum = e;
#pragma unroll
    for (int d = 1; d < 16; d <<= 1) sum += __shfl_xor(sum, d, 32);
    attn_s[wave][i + 8 * (lane >> 4)][m] = e / sum;  // row, col of 16x16 attn
  }

  // wave-local LDS RAW: explicit CDNA5 DS counter wait (data is per-wave, no block barrier needed)
  asm volatile("s_wait_dscnt 0" ::: "memory");

  // out = attn @ vh : 4 strips of 16 cols, K=16 in 4 steps of V_WMMA_F32_16X16X4_F32
  float* out = Ctx + (size_t)tok * DM;
#pragma unroll
  for (int t = 0; t < 4; ++t) {
    v8f o = {};
    const int d = t * 16 + m;
#pragma unroll
    for (int ks = 0; ks < NH; ks += 4) {
      v2f a = { attn_s[wave][m][ks + kb], attn_s[wave][m][ks + kb + 1] };
      v2f b = { v[(ks + kb) * HD + d], v[(ks + kb + 1) * HD + d] };
      o = __builtin_amdgcn_wmma_f32_16x16x4_f32(false, a, false, b, (short)0, o, false, false);
    }
#pragma unroll
    for (int i = 0; i < 8; ++i) {
      out[(size_t)(i + 8 * (lane >> 4)) * HD + d] = o[i];
    }
  }
}

extern "C" void kernel_launch(void* const* d_in, const int* in_sizes, int n_in,
                              void* d_out, int out_size, void* d_ws, size_t ws_size,
                              hipStream_t stream) {
  const float* q  = (const float*)d_in[0];
  const float* k  = (const float*)d_in[1];
  const float* v  = (const float*)d_in[2];
  const float* Wq = (const float*)d_in[3];
  const float* bq = (const float*)d_in[4];
  const float* Wk = (const float*)d_in[5];
  const float* bk = (const float*)d_in[6];
  const float* Wv = (const float*)d_in[7];
  const float* bv = (const float*)d_in[8];
  const float* Wo = (const float*)d_in[9];
  const float* bo = (const float*)d_in[10];
  float* out = (float*)d_out;

  const size_t ND = (size_t)N_TOK * DM;   // 16M floats
  const size_t WE = (size_t)DM * DM;      // 1M elems per weight matrix
  float* Qh  = (float*)d_ws;
  float* Kh  = Qh + ND;
  float* Vh  = Kh + ND;
  float* Ctx = Vh + ND;
  unsigned short* wb = (unsigned short*)(Ctx + ND);
  unsigned short* WqHi = wb + 0 * WE;  unsigned short* WqLo = wb + 1 * WE;
  unsigned short* WkHi = wb + 2 * WE;  unsigned short* WkLo = wb + 3 * WE;
  unsigned short* WvHi = wb + 4 * WE;  unsigned short* WvLo = wb + 5 * WE;
  unsigned short* WoHi = wb + 6 * WE;  unsigned short* WoLo = wb + 7 * WE;

  // 1) split weights into bf16 hi/lo once
  const int cvt_blocks = (int)(WE / (256 * 8));  // 512
  cvt_f32_bf16_hilo<<<cvt_blocks, 256, 0, stream>>>(Wq, WqHi, WqLo);
  cvt_f32_bf16_hilo<<<cvt_blocks, 256, 0, stream>>>(Wk, WkHi, WkLo);
  cvt_f32_bf16_hilo<<<cvt_blocks, 256, 0, stream>>>(Wv, WvHi, WvLo);
  cvt_f32_bf16_hilo<<<cvt_blocks, 256, 0, stream>>>(Wo, WoHi, WoLo);

  // 2) projections
  dim3 gemm_grid(N_TOK / 128, DM / 64);  // 128x16 blocks, 256 threads (8 waves) each
  gemm_xwt_bias_f32<<<gemm_grid, 256, 0, stream>>>(q, WqHi, WqLo, bq, Qh);
  gemm_xwt_bias_f32<<<gemm_grid, 256, 0, stream>>>(k, WkHi, WkLo, bk, Kh);
  gemm_xwt_bias_f32<<<gemm_grid, 256, 0, stream>>>(v, WvHi, WvLo, bv, Vh);

  // 3) per-token attention over heads
  attn_over_heads<<<dim3(N_TOK / 8), 256, 0, stream>>>(Qh, Kh, Vh, Ctx);

  // 4) output projection
  gemm_xwt_bias_f32<<<gemm_grid, 256, 0, stream>>>(Ctx, WoHi, WoLo, bo, out);
}
